// HINormer_2388001816792
// MI455X (gfx1250) — matrix operand order repaired and checked by reading, not compiled, not run on hardware
//
#include <hip/hip_runtime.h>
#include <hip/hip_bf16.h>

// ---------------- problem constants (from reference) ----------------
#define NA_N 50000
#define NP_N 100000
#define NS_N 1000
#define DIN  128
#define HID  64
#define HEADS 4
#define OUTD 64
#define HC1  (HEADS * HID)   // 256
#define E_AP 300000
#define E_PS 150000

typedef __bf16 bf16;
typedef bf16  v16bf __attribute__((ext_vector_type(16)));
typedef bf16  v8bf  __attribute__((ext_vector_type(8)));
typedef float v8f   __attribute__((ext_vector_type(8)));

// ---------------- elementwise utility kernels ----------------
__global__ void hn_fill_u32(unsigned* __restrict__ p, unsigned v, long long n) {
  long long i = (long long)blockIdx.x * blockDim.x + threadIdx.x;
  if (i < n) p[i] = v;
}

__global__ void hn_cvt_bf16(const float* __restrict__ in, bf16* __restrict__ out, long long n) {
  long long i = (long long)blockIdx.x * blockDim.x + threadIdx.x;
  if (i < n) out[i] = (bf16)in[i];
}

// W[K x N] f32 (row-major)  ->  Wt[N x K] bf16 (row-major, i.e. B transposed)
__global__ void hn_wT_bf16(const float* __restrict__ W, bf16* __restrict__ Wt, int K, int N) {
  long long i = (long long)blockIdx.x * blockDim.x + threadIdx.x;
  if (i >= (long long)K * N) return;
  int k = (int)(i / N);
  int n = (int)(i - (long long)k * N);
  Wt[(size_t)n * K + k] = (bf16)W[i];
}

// ---------------- WMMA bf16 GEMM: C[MxN] = A[MxK] @ W[KxN] + bias ----------------
// A: bf16 row-major, Wt: bf16 [N x K] (transposed weight), bias: f32[N] (nullable)
// Each wave computes a 16x64 strip; 4 waves per block -> 64 rows/block.
// Requires N % 64 == 0 and K % 32 == 0 (true for all GEMMs here).
__global__ void __launch_bounds__(128)
hn_gemm_bf16(const bf16* __restrict__ A, const bf16* __restrict__ Wt,
             const float* __restrict__ bias, float* __restrict__ C,
             int M, int N, int K) {
  const int wave = threadIdx.x >> 5;
  const int lane = threadIdx.x & 31;
  const int m16  = lane & 15;
  const int half = lane >> 4;

  const int row0 = (blockIdx.x * 4 + wave) * 16;   // wave-uniform
  const int col0 = blockIdx.y * 64;
  if (row0 >= M) return;                           // uniform per wave: EXEC stays all-ones

  int rA = row0 + m16;                              // clamp tail rows (garbage rows masked at store)
  if (rA > M - 1) rA = M - 1;

  v8f acc[4] = {};

  for (int k0 = 0; k0 < K; k0 += 32) {
    // A fragment: lane (m, half) holds K in
    // [half*16, half*16+16) (contiguous, matching B)
    union { v16bf v; v8bf h[2]; } a;
    const bf16* ap = A + (size_t)rA * K + k0;
    a.h[0] = *(const v8bf*)(ap + half * 16);
    a.h[1] = *(const v8bf*)(ap + half * 16 + 8);

#pragma unroll
    for (int t = 0; t < 4; ++t) {
      // B fragment: lane (n=m16, half) holds K in [half*16, half*16+16),
      // contiguous in the transposed weight -> single 32B vector load.
      const int n = col0 + t * 16 + m16;
      v16bf b = *(const v16bf*)(Wt + (size_t)n * K + k0 + half * 16);
      acc[t] = __builtin_amdgcn_wmma_f32_16x16x32_bf16(
          /*neg_a=*/false, a.v, /*neg_b=*/false, b,
          /*c_mod=*/(short)0, acc[t], /*reuse_a=*/false, /*reuse_b=*/false);
    }
  }

  // C/D layout: VGPR v, lanes 0-15 -> M=v ; lanes 16-31 -> M=8+v ; N = lane&15
#pragma unroll
  for (int t = 0; t < 4; ++t) {
    const int n = col0 + t * 16 + m16;
    const float bn = bias ? bias[n] : 0.f;
#pragma unroll
    for (int v = 0; v < 8; ++v) {
      const int r = row0 + half * 8 + v;
      if (r < M) C[(size_t)r * N + n] = acc[t][v] + bn;
    }
  }
}

// ---------------- GATv2 edge kernels ----------------
// logits[e,h] = sum_c att[h,c] * leaky_relu(xi + xj, 0.2); also segment-max via
// order-preserving flipped-uint atomicMax.
__global__ void hn_gat_logits(const float* __restrict__ hsl, const float* __restrict__ hdr,
                              const float* __restrict__ att, const int* __restrict__ src,
                              const int* __restrict__ dst, float* __restrict__ logits,
                              unsigned* __restrict__ mx, int E, int H, int C) {
  int idx = blockIdx.x * blockDim.x + threadIdx.x;
  if (idx >= E * H) return;
  const int e = idx / H, h = idx - e * H;
  const int HC = H * C;
  const float* xj = hsl + (size_t)src[e] * HC + h * C;
  const float* xi = hdr + (size_t)dst[e] * HC + h * C;
  const float* ah = att + h * C;
  float s = 0.f;
#pragma unroll 8
  for (int c = 0; c < C; ++c) {
    float v = xi[c] + xj[c];
    v = v > 0.f ? v : 0.2f * v;
    s += ah[c] * v;
  }
  logits[idx] = s;
  unsigned u = __float_as_uint(s);
  u = (u & 0x80000000u) ? ~u : (u | 0x80000000u);
  atomicMax(mx + (size_t)dst[e] * H + h, u);
}

// ex = exp(logit - max[dst]); den[dst] += ex   (logits buffer rewritten in place)
__global__ void hn_gat_expsum(const int* __restrict__ dst, const unsigned* __restrict__ mx,
                              float* __restrict__ exlog, float* __restrict__ den,
                              int E, int H) {
  int idx = blockIdx.x * blockDim.x + threadIdx.x;
  if (idx >= E * H) return;
  const int e = idx / H, h = idx - e * H;
  const unsigned u = mx[(size_t)dst[e] * H + h];
  const float m = __uint_as_float((u & 0x80000000u) ? (u ^ 0x80000000u) : ~u);
  const float ex = expf(exlog[idx] - m);
  exlog[idx] = ex;
  atomicAdd(den + (size_t)dst[e] * H + h, ex);
}

// out[dst, j] += xj[src, j] * (ex / den[dst])
__global__ void hn_gat_scatter(const float* __restrict__ hsl, const int* __restrict__ src,
                               const int* __restrict__ dst, const float* __restrict__ ex,
                               const float* __restrict__ den, float* __restrict__ out,
                               int E, int H, int C) {
  const int HC = H * C;
  long long idx = (long long)blockIdx.x * blockDim.x + threadIdx.x;
  if (idx >= (long long)E * HC) return;
  const int e = (int)(idx / HC);
  const int j = (int)(idx - (long long)e * HC);
  const int h = j / C;
  const float alpha = ex[(size_t)e * H + h] / den[(size_t)dst[e] * H + h];
  atomicAdd(out + (size_t)dst[e] * HC + j, hsl[(size_t)src[e] * HC + j] * alpha);
}

// out_bf16 = ELU( scale * (in + b1 + (b2?b2:0)) )   (HeteroConv mean + bias + ELU)
__global__ void hn_combine_elu(const float* __restrict__ in, const float* __restrict__ b1,
                               const float* __restrict__ b2, float scale,
                               bf16* __restrict__ outbf, long long N, int HC) {
  long long idx = (long long)blockIdx.x * blockDim.x + threadIdx.x;
  if (idx >= N * HC) return;
  const int j = (int)(idx % HC);
  const float bb = b2 ? b2[j] : 0.f;
  float v = scale * (in[idx] + b1[j] + bb);
  v = v > 0.f ? v : (expf(v) - 1.f);
  outbf[idx] = (bf16)v;
}

// y = scale*(x + b1 + b2), then LayerNorm(y)*g + b  ; one thread per row (D=64)
__global__ void hn_finalize_ln(const float* __restrict__ in, const float* __restrict__ b1,
                               const float* __restrict__ b2, float scale,
                               const float* __restrict__ g, const float* __restrict__ bln,
                               float* __restrict__ out, int N, int D) {
  int r = blockIdx.x * blockDim.x + threadIdx.x;
  if (r >= N) return;
  const float* x = in + (size_t)r * D;
  float mu = 0.f;
  for (int i = 0; i < D; ++i) {
    const float bb = b2 ? b2[i] : 0.f;
    mu += scale * (x[i] + b1[i] + bb);
  }
  mu /= (float)D;
  float var = 0.f;
  for (int i = 0; i < D; ++i) {
    const float bb = b2 ? b2[i] : 0.f;
    const float d = scale * (x[i] + b1[i] + bb) - mu;
    var += d * d;
  }
  var /= (float)D;
  const float inv = rsqrtf(var + 1e-5f);
  for (int i = 0; i < D; ++i) {
    const float bb = b2 ? b2[i] : 0.f;
    out[(size_t)r * D + i] = (scale * (x[i] + b1[i] + bb) - mu) * inv * g[i] + bln[i];
  }
}

// ---------------- host orchestration ----------------
static inline dim3 grid1(long long n, int b) { return dim3((unsigned)((n + b - 1) / b)); }

extern "C" void kernel_launch(void* const* d_in, const int* in_sizes, int n_in,
                              void* d_out, int out_size, void* d_ws, size_t ws_size,
                              hipStream_t stream) {
  (void)in_sizes; (void)n_in; (void)out_size; (void)ws_size;

  // Input flattening (jax tree leaves, dict keys sorted):
  // 0..2: x_author, x_paper, x_subject
  // params: conv1{about,includes,writes,written_by}{att,bias,llW,llb,lrW,lrb} = 3..26
  //         conv2 same = 27..50 ; ln{b,g} = 51,52 ; proj{author,paper,subject}{W,b} = 53..58
  // 59..66: ew_src, ew_dst, ewb_src, ewb_dst, ea_src, ea_dst, ei_src, ei_dst
  const float* xin[3] = {(const float*)d_in[0], (const float*)d_in[1], (const float*)d_in[2]};
  const int nodeN[3]  = {NA_N, NP_N, NS_N};      // 0=author, 1=paper, 2=subject
  const int C1B = 3, C2B = 27, LNB = 51, LNG = 52, PRJ = 53;
  // relation order: 0=about(p->s), 1=includes(s->p), 2=writes(a->p), 3=written_by(p->a)
  const int RSRC[4] = {1, 2, 0, 1};
  const int RDST[4] = {2, 1, 1, 0};
  const int RE[4]   = {E_PS, E_PS, E_AP, E_AP};
  const int ESI[4]  = {63, 65, 59, 61};
  const int EDI[4]  = {64, 66, 60, 62};

  // ---- workspace bump allocator ----
  char* ws = (char*)d_ws;
  size_t off = 0;
  auto alloc = [&](size_t bytes) -> void* {
    void* p = ws + off;
    off += (bytes + 255) & ~(size_t)255;
    return p;
  };
  float* hfeat[3]; bf16* hbf[3]; float* out1[3]; bf16* h1bf[3]; float* out2[3];
  for (int t = 0; t < 3; ++t) hfeat[t] = (float*)alloc((size_t)nodeN[t] * HID * 4);
  for (int t = 0; t < 3; ++t) hbf[t]   = (bf16*)alloc((size_t)nodeN[t] * HID * 2);
  for (int t = 0; t < 3; ++t) out1[t]  = (float*)alloc((size_t)nodeN[t] * HC1 * 4);
  for (int t = 0; t < 3; ++t) h1bf[t]  = (bf16*)alloc((size_t)nodeN[t] * HC1 * 2);
  for (int t = 0; t < 3; ++t) out2[t]  = (float*)alloc((size_t)nodeN[t] * OUTD * 4);
  bf16*     wt  = (bf16*)alloc((size_t)256 * 256 * 2);
  unsigned* mx  = (unsigned*)alloc((size_t)NP_N * HEADS * 4);
  float*    den = (float*)alloc((size_t)NP_N * HEADS * 4);
  float*    exb = (float*)alloc((size_t)E_AP * HEADS * 4);
  bf16*     xbf = (bf16*)alloc((size_t)NP_N * DIN * 2);
  float*    hsl = (float*)alloc((size_t)NP_N * HC1 * 4);
  float*    hdr = (float*)alloc((size_t)NP_N * HC1 * 4);

  // ---- Phase A: type projections (128 -> 64), keep f32 + bf16 copies ----
  for (int t = 0; t < 3; ++t) {
    const float* W = (const float*)d_in[PRJ + 2 * t];
    const float* b = (const float*)d_in[PRJ + 2 * t + 1];
    hn_wT_bf16<<<grid1((long long)DIN * HID, 256), 256, 0, stream>>>(W, wt, DIN, HID);
    const long long nx = (long long)nodeN[t] * DIN;
    hn_cvt_bf16<<<grid1(nx, 256), 256, 0, stream>>>(xin[t], xbf, nx);
    hn_gemm_bf16<<<dim3((nodeN[t] + 63) / 64, HID / 64), 128, 0, stream>>>(
        xbf, wt, b, hfeat[t], nodeN[t], HID, DIN);
    const long long nh = (long long)nodeN[t] * HID;
    hn_cvt_bf16<<<grid1(nh, 256), 256, 0, stream>>>(hfeat[t], hbf[t], nh);
  }

  // ---- Phase B: conv1 (H=4, C=64) ----
  for (int t = 0; t < 3; ++t) {
    const long long n = (long long)nodeN[t] * HC1;
    hn_fill_u32<<<grid1(n, 256), 256, 0, stream>>>((unsigned*)out1[t], 0u, n);
  }
  for (int r = 0; r < 4; ++r) {
    const int s = RSRC[r], dt = RDST[r], E = RE[r];
    const int base = C1B + 6 * r;
    const float* att = (const float*)d_in[base];
    const float* llW = (const float*)d_in[base + 2];
    const float* llb = (const float*)d_in[base + 3];
    const float* lrW = (const float*)d_in[base + 4];
    const float* lrb = (const float*)d_in[base + 5];
    const int* esrc = (const int*)d_in[ESI[r]];
    const int* edst = (const int*)d_in[EDI[r]];

    hn_wT_bf16<<<grid1((long long)HID * HC1, 256), 256, 0, stream>>>(llW, wt, HID, HC1);
    hn_gemm_bf16<<<dim3((nodeN[s] + 63) / 64, HC1 / 64), 128, 0, stream>>>(
        hbf[s], wt, llb, hsl, nodeN[s], HC1, HID);
    hn_wT_bf16<<<grid1((long long)HID * HC1, 256), 256, 0, stream>>>(lrW, wt, HID, HC1);
    hn_gemm_bf16<<<dim3((nodeN[dt] + 63) / 64, HC1 / 64), 128, 0, stream>>>(
        hbf[dt], wt, lrb, hdr, nodeN[dt], HC1, HID);

    const long long nd = (long long)nodeN[dt] * HEADS;
    hn_fill_u32<<<grid1(nd, 256), 256, 0, stream>>>(mx, 0u, nd);
    hn_fill_u32<<<grid1(nd, 256), 256, 0, stream>>>((unsigned*)den, 0u, nd);
    const long long nEH = (long long)E * HEADS;
    hn_gat_logits<<<grid1(nEH, 256), 256, 0, stream>>>(hsl, hdr, att, esrc, edst, exb, mx, E, HEADS, HID);
    hn_gat_expsum<<<grid1(nEH, 256), 256, 0, stream>>>(edst, mx, exb, den, E, HEADS);
    hn_gat_scatter<<<grid1((long long)E * HC1, 256), 256, 0, stream>>>(
        hsl, esrc, edst, exb, den, out1[dt], E, HEADS, HID);
  }
  // mean over relations per dst type + bias + ELU, emit bf16 for conv2 GEMMs
  hn_combine_elu<<<grid1((long long)NA_N * HC1, 256), 256, 0, stream>>>(
      out1[0], (const float*)d_in[C1B + 6 * 3 + 1], nullptr, 1.f, h1bf[0], NA_N, HC1);
  hn_combine_elu<<<grid1((long long)NP_N * HC1, 256), 256, 0, stream>>>(
      out1[1], (const float*)d_in[C1B + 6 * 2 + 1], (const float*)d_in[C1B + 6 * 1 + 1],
      0.5f, h1bf[1], NP_N, HC1);
  hn_combine_elu<<<grid1((long long)NS_N * HC1, 256), 256, 0, stream>>>(
      out1[2], (const float*)d_in[C1B + 6 * 0 + 1], nullptr, 1.f, h1bf[2], NS_N, HC1);

  // ---- Phase C: conv2 (H=1, C=64, K=256) ----
  for (int t = 0; t < 3; ++t) {
    const long long n = (long long)nodeN[t] * OUTD;
    hn_fill_u32<<<grid1(n, 256), 256, 0, stream>>>((unsigned*)out2[t], 0u, n);
  }
  for (int r = 0; r < 4; ++r) {
    const int s = RSRC[r], dt = RDST[r], E = RE[r];
    const int base = C2B + 6 * r;
    const float* att = (const float*)d_in[base];
    const float* llW = (const float*)d_in[base + 2];
    const float* llb = (const float*)d_in[base + 3];
    const float* lrW = (const float*)d_in[base + 4];
    const float* lrb = (const float*)d_in[base + 5];
    const int* esrc = (const int*)d_in[ESI[r]];
    const int* edst = (const int*)d_in[EDI[r]];

    hn_wT_bf16<<<grid1((long long)HC1 * OUTD, 256), 256, 0, stream>>>(llW, wt, HC1, OUTD);
    hn_gemm_bf16<<<dim3((nodeN[s] + 63) / 64, OUTD / 64), 128, 0, stream>>>(
        h1bf[s], wt, llb, hsl, nodeN[s], OUTD, HC1);
    hn_wT_bf16<<<grid1((long long)HC1 * OUTD, 256), 256, 0, stream>>>(lrW, wt, HC1, OUTD);
    hn_gemm_bf16<<<dim3((nodeN[dt] + 63) / 64, OUTD / 64), 128, 0, stream>>>(
        h1bf[dt], wt, lrb, hdr, nodeN[dt], OUTD, HC1);

    const long long nd = (long long)nodeN[dt];
    hn_fill_u32<<<grid1(nd, 256), 256, 0, stream>>>(mx, 0u, nd);
    hn_fill_u32<<<grid1(nd, 256), 256, 0, stream>>>((unsigned*)den, 0u, nd);
    const long long nE = (long long)E;
    hn_gat_logits<<<grid1(nE, 256), 256, 0, stream>>>(hsl, hdr, att, esrc, edst, exb, mx, E, 1, OUTD);
    hn_gat_expsum<<<grid1(nE, 256), 256, 0, stream>>>(edst, mx, exb, den, E, 1);
    hn_gat_scatter<<<grid1((long long)E * OUTD, 256), 256, 0, stream>>>(
        hsl, esrc, edst, exb, den, out2[dt], E, 1, OUTD);
  }

  // ---- Finalize: bias (+mean for paper) + LayerNorm -> d_out (author, paper, subject) ----
  const float* g   = (const float*)d_in[LNG];
  const float* bln = (const float*)d_in[LNB];
  float* outp = (float*)d_out;
  hn_finalize_ln<<<grid1(NA_N, 256), 256, 0, stream>>>(
      out2[0], (const float*)d_in[C2B + 6 * 3 + 1], nullptr, 1.f, g, bln,
      outp, NA_N, OUTD);
  hn_finalize_ln<<<grid1(NP_N, 256), 256, 0, stream>>>(
      out2[1], (const float*)d_in[C2B + 6 * 2 + 1], (const float*)d_in[C2B + 6 * 1 + 1],
      0.5f, g, bln, outp + (size_t)NA_N * OUTD, NP_N, OUTD);
  hn_finalize_ln<<<grid1(NS_N, 256), 256, 0, stream>>>(
      out2[2], (const float*)d_in[C2B + 6 * 0 + 1], nullptr, 1.f, g, bln,
      outp + (size_t)(NA_N + NP_N) * OUTD, NS_N, OUTD);
}
